// _PointnetSAModuleBase_21973052686362
// MI455X (gfx1250) — matrix operand order repaired and checked by reading, not compiled, not run on hardware
//
#include <hip/hip_runtime.h>

// ---------------- constants ----------------
constexpr int BATCH = 8;
constexpr int NPTS  = 16384;
constexpr int NP    = 512;    // npoint
constexpr int KNN   = 32;
constexpr int CF    = 128;
constexpr int CO    = 128;
constexpr int CG    = 137;    // 9 + 128 grouped channels
constexpr int CGP   = 160;    // padded K for WMMA (5 * 32)
constexpr int NG    = 32;     // groupnorm groups
constexpr float GEPS   = 1e-5f;
constexpr float RAD2   = 0.1f * 0.1f;
constexpr float QKSCALE = 0.088388347648318447f; // 128^-0.5
constexpr float INVCNT  = 1.0f / 65536.0f;       // 4 channels * 16384 spatial

typedef __attribute__((ext_vector_type(16))) __bf16 v16bf;
typedef __attribute__((ext_vector_type(8)))  float  v8f;

#if defined(__HIP_DEVICE_COMPILE__) && __has_builtin(__builtin_amdgcn_tensor_load_to_lds)
#define USE_TDM 1
#else
#define USE_TDM 0
#endif

#if USE_TDM
typedef unsigned u32x4 __attribute__((ext_vector_type(4)));
typedef int      i32x8 __attribute__((ext_vector_type(8)));
typedef int      i32x4 __attribute__((ext_vector_type(4)));
#endif

// ---------------- TDM stage: (128 x 16) f32 panel (row stride NPTS) -> LDS ----
// LDS layout: row stride 17 dwords (TDM pad: 1 dword after every 16) to avoid
// bank conflicts between lane halves when reading B fragments.
static __device__ __forceinline__ void stage_panel128(float* lds,
                                                      const float* src) {
#if USE_TDM
  const unsigned lds_off =
      (unsigned)(unsigned long long)(__attribute__((address_space(3))) float*)lds;
  const unsigned long long ga = (unsigned long long)(const void*)src;
  u32x4 g0;
  g0[0] = 1u;                                   // count=1, user descriptor
  g0[1] = lds_off;                              // lds_addr
  g0[2] = (unsigned)ga;                         // global_addr[31:0]
  g0[3] = (unsigned)(ga >> 32) | 0x80000000u;   // global_addr[56:32] | type=2
  i32x8 g1;
  g1[0] = (2 << 16) | (1 << 20) | (3 << 22);    // data_size=4B, pad_en, every 16DW pad 1DW
  g1[1] = (int)(((unsigned)NPTS & 0xFFFFu) << 16);   // tensor_dim0 low16
  g1[2] = (int)((((unsigned)NPTS >> 16) & 0xFFFFu)   // tensor_dim0 high16
              | (((unsigned)CF & 0xFFFFu) << 16));   // tensor_dim1 low16
  g1[3] = (int)(16u << 16);                     // tensor_dim1 hi16=0 | tile_dim0=16
  g1[4] = CF;                                   // tile_dim1=128, tile_dim2=0
  g1[5] = NPTS;                                 // tensor_dim0_stride low32
  g1[6] = 0;
  g1[7] = 0;
  i32x4 z4 = {0, 0, 0, 0};
  i32x8 z8 = {0, 0, 0, 0, 0, 0, 0, 0};
  __builtin_amdgcn_tensor_load_to_lds(g0, g1, z4, z4, z8, 0);
  __builtin_amdgcn_s_wait_tensorcnt(0);
#else
  const int lane = threadIdx.x & 31;
  for (int i = lane; i < CF * 16; i += 32) {
    const int r = i >> 4, c = i & 15;
    lds[r * 17 + c] = src[(size_t)r * NPTS + c];
  }
#endif
  __syncthreads();
}

// ---------------- WMMA tile helper (bf16 in, f32 accum) ----------------
// One wave computes one 16x16 f32 tile, looping K in steps of 32.
// A fragment per ISA 16-bit A 16x32 layout; B per 32x16 layout; D per 32-bit C/D.
template <int KSTEPS, typename GA, typename GB>
static __device__ __forceinline__ v8f wmma_loop(GA ga, GB gb) {
  v8f acc = {0.f,0.f,0.f,0.f,0.f,0.f,0.f,0.f};
  const int lane = threadIdx.x & 31;
  const int half = lane >> 4;
  const int mn   = lane & 15;
  #pragma unroll
  for (int ks = 0; ks < KSTEPS; ++ks) {
    const int kb = ks << 5;
    v16bf a, bv;
    #pragma unroll
    for (int j = 0; j < 8; ++j) {
      const int ka = ((j < 4) ? (2*j) : (16 + 2*(j-4))) + half*8;
      a[2*j]   = (__bf16)ga(mn, kb + ka);
      a[2*j+1] = (__bf16)ga(mn, kb + ka + 1);
      const int kbb = ((j < 4) ? (2*j) : (16 + 2*(j-4))) + half*8;
      bv[2*j]   = (__bf16)gb(kb + kbb,     mn);
      bv[2*j+1] = (__bf16)gb(kb + kbb + 1, mn);
    }
    acc = __builtin_amdgcn_wmma_f32_16x16x32_bf16(false, a, false, bv,
                                                  (short)0, acc, false, false);
  }
  return acc;
}

// wave-level groupnorm partial-sum accumulation for a 16-channel tile
static __device__ __forceinline__ void group_accum(float* gsum, float* gss,
                                                   int b, int mtile,
                                                   const float* vals,
                                                   int half, int lane) {
  float s0=0.f,q0=0.f,s1=0.f,q1=0.f;
  #pragma unroll
  for (int r = 0; r < 8; ++r) {
    float v = vals[r];
    if (r < 4) { s0 += v; q0 += v*v; } else { s1 += v; q1 += v*v; }
  }
  #pragma unroll
  for (int o = 1; o < 16; o <<= 1) {
    s0 += __shfl_xor(s0, o, 32); q0 += __shfl_xor(q0, o, 32);
    s1 += __shfl_xor(s1, o, 32); q1 += __shfl_xor(q1, o, 32);
  }
  if ((lane & 15) == 0) {
    int gb = mtile*4 + half*2;
    atomicAdd(&gsum[b*NG + gb],     s0); atomicAdd(&gss[b*NG + gb],     q0);
    atomicAdd(&gsum[b*NG + gb + 1], s1); atomicAdd(&gss[b*NG + gb + 1], q1);
  }
}

// ---------------- small MLPs: t and condition embeddings ----------------
__global__ void k_embed(const float* temb_in, const float* tw1, const float* tb1,
                        const float* tw2, const float* tb2,
                        const float* cond_in, const float* cw1, const float* cb1,
                        const float* cw2, const float* cb2,
                        float* tout, float* cout_) {
  __shared__ float h[128];
  const int b = blockIdx.x, o = threadIdx.x;
  float a = tb1[o];
  for (int i = 0; i < 512; ++i) a += temb_in[b*512 + i] * tw1[o*512 + i];
  h[o] = fmaxf(a, 0.f);
  __syncthreads();
  float a2 = tb2[o];
  for (int j = 0; j < 128; ++j) a2 += h[j] * tw2[o*128 + j];
  tout[b*128 + o] = fmaxf(a2, 0.f);

  a = cb1[o];
  for (int i = 0; i < 1024; ++i) a += cond_in[b*1024 + i] * cw1[o*1024 + i];
  __syncthreads();
  h[o] = fmaxf(a, 0.f);
  __syncthreads();
  a2 = cb2[o];
  for (int j = 0; j < 128; ++j) a2 += h[j] * cw2[o*128 + j];
  cout_[b*128 + o] = fmaxf(a2, 0.f);
}

// ---------------- furthest point sampling (one block per batch) ----------------
__global__ void k_fps(const float* xyz, int* fidx, float* newxyz) {
  const int b = blockIdx.x, t = threadIdx.x;      // 512 threads
  constexpr int PT = NPTS / 512;                  // 32 points per thread
  __shared__ float sv[512];
  __shared__ int   si[512];
  const float* xb = xyz + (size_t)b * NPTS * 3;
  const float p0x = xb[0], p0y = xb[1], p0z = xb[2];
  float dist[PT];
  #pragma unroll
  for (int j = 0; j < PT; ++j) {
    int p = t + 512*j;
    float dx = xb[p*3+0]-p0x, dy = xb[p*3+1]-p0y, dz = xb[p*3+2]-p0z;
    dist[j] = dx*dx + dy*dy + dz*dz;
  }
  if (t == 0) {
    fidx[b*NP] = 0;
    newxyz[(size_t)b*NP*3 + 0] = p0x;
    newxyz[(size_t)b*NP*3 + 1] = p0y;
    newxyz[(size_t)b*NP*3 + 2] = p0z;
  }
  for (int it = 1; it < NP; ++it) {
    float best = -1.f; int bid = 0;
    #pragma unroll
    for (int j = 0; j < PT; ++j)
      if (dist[j] > best) { best = dist[j]; bid = t + 512*j; }
    sv[t] = best; si[t] = bid;
    __syncthreads();
    for (int s = 256; s > 0; s >>= 1) {
      if (t < s && sv[t+s] > sv[t]) { sv[t] = sv[t+s]; si[t] = si[t+s]; }
      __syncthreads();
    }
    const int w = si[0];
    const float wx = xb[w*3+0], wy = xb[w*3+1], wz = xb[w*3+2];
    if (t == 0) {
      fidx[b*NP + it] = w;
      newxyz[((size_t)b*NP + it)*3 + 0] = wx;
      newxyz[((size_t)b*NP + it)*3 + 1] = wy;
      newxyz[((size_t)b*NP + it)*3 + 2] = wz;
    }
    #pragma unroll
    for (int j = 0; j < PT; ++j) {
      int p = t + 512*j;
      float dx = xb[p*3+0]-wx, dy = xb[p*3+1]-wy, dz = xb[p*3+2]-wz;
      dist[j] = fminf(dist[j], dx*dx + dy*dy + dz*dz);
    }
    __syncthreads();
  }
}

// ---------------- ball query (one wave per center) ----------------
__global__ void k_ball(const float* xyz, const float* newxyz, int* idxb, int* cntb) {
  const int wv = threadIdx.x >> 5, lane = threadIdx.x & 31;
  const int ci = blockIdx.x * 8 + wv;             // [0, BATCH*NP)
  const int b = ci >> 9, n = ci & 511;
  const float cx = newxyz[((size_t)b*NP + n)*3 + 0];
  const float cy = newxyz[((size_t)b*NP + n)*3 + 1];
  const float cz = newxyz[((size_t)b*NP + n)*3 + 2];
  const float* xb = xyz + (size_t)b * NPTS * 3;
  int cnt = 0, firstv = -1;
  for (int j = 0; j < NPTS && cnt < KNN; j += 32) {
    const int p = j + lane;
    const float dx = xb[p*3+0]-cx, dy = xb[p*3+1]-cy, dz = xb[p*3+2]-cz;
    const bool in = (dx*dx + dy*dy + dz*dz) < RAD2;
    const unsigned bal = (unsigned)__ballot(in);
    const int rank = __popc(bal & ((1u << lane) - 1u));
    const int slot = cnt + rank;
    if (in && slot < KNN) {
      idxb[((size_t)(b*NP + n))*KNN + slot] = p;
      if (slot == 0) firstv = p;
    }
    cnt += __popc(bal);
  }
  const int c = cnt < KNN ? cnt : KNN;
  #pragma unroll
  for (int o = 16; o > 0; o >>= 1) firstv = max(firstv, __shfl_xor(firstv, o, 32));
  const int first = (firstv >= 0) ? firstv : (NPTS - 1);
  if (lane >= c) idxb[((size_t)(b*NP + n))*KNN + lane] = first;
  if (lane == 0) cntb[b*NP + n] = c;
}

// ---------------- GEMM: fm  (128 x 256 @ 256 x N), TDM-staged B panel ----------
__global__ void k_gemm_fm(const float* W, const float* bias,
                          const float* features, const float* temb,
                          float* out, float* gsum, float* gss) {
  __shared__ float tb_lds[CF * 17];
  const int nb = blockIdx.x * 16, mtile = blockIdx.y, b = blockIdx.z;
  const int mb = mtile * 16;
  const float* fb = features + (size_t)b * CF * NPTS;
  const float* tb = temb + b * 128;
  stage_panel128(tb_lds, fb + nb);
  auto ga = [&](int m, int k) { return W[(mb + m)*256 + k]; };
  auto gb = [&](int k, int nl) {
    return (k < CF) ? tb_lds[k*17 + nl] : tb[k - CF];
  };
  v8f acc = wmma_loop<8>(ga, gb);
  const int lane = threadIdx.x & 31, half = lane >> 4, nl = lane & 15;
  float vals[8];
  #pragma unroll
  for (int r = 0; r < 8; ++r) {
    const int M = r + half*8;
    const float v = acc[r] + bias[mb + M];
    out[(size_t)b*CO*NPTS + (size_t)(mb + M)*NPTS + (nb + nl)] = v;
    vals[r] = v;
  }
  group_accum(gsum, gss, b, mtile, vals, half, lane);
}

// ---------------- GEMM: sm  (gn1+relu folded into LDS reads of pre1) -----------
__global__ void k_gemm_sm(const float* W, const float* bias,
                          const float* pre1, const float* st1,
                          const float* g1, const float* bb1,
                          const float* cemb,
                          float* out, float* gsum, float* gss) {
  __shared__ float tb_lds[CO * 17];
  const int nb = blockIdx.x * 16, mtile = blockIdx.y, b = blockIdx.z;
  const int mb = mtile * 16;
  const float* p1 = pre1 + (size_t)b * CO * NPTS;
  const float* cb = cemb + b * 128;
  stage_panel128(tb_lds, p1 + nb);
  auto ga = [&](int m, int k) { return W[(mb + m)*256 + k]; };
  auto gb = [&](int k, int nl) {
    if (k < CO) {
      const int s = (b*NG + (k >> 2)) * 2;
      const float x = tb_lds[k*17 + nl];
      return fmaxf((x - st1[s]) * st1[s+1] * g1[k] + bb1[k], 0.f);
    }
    return cb[k - CO];
  };
  v8f acc = wmma_loop<8>(ga, gb);
  const int lane = threadIdx.x & 31, half = lane >> 4, nl = lane & 15;
  float vals[8];
  #pragma unroll
  for (int r = 0; r < 8; ++r) {
    const int M = r + half*8;
    const float v = acc[r] + bias[mb + M];
    out[(size_t)b*CO*NPTS + (size_t)(mb + M)*NPTS + (nb + nl)] = v;
    vals[r] = v;
  }
  group_accum(gsum, gss, b, mtile, vals, half, lane);
}

// ---------------- GEMM: rc  (128 x 128 @ features), TDM-staged B panel --------
__global__ void k_gemm_rc(const float* W, const float* bias,
                          const float* features,
                          float* out, float* gsum, float* gss) {
  __shared__ float tb_lds[CF * 17];
  const int nb = blockIdx.x * 16, mtile = blockIdx.y, b = blockIdx.z;
  const int mb = mtile * 16;
  const float* fb = features + (size_t)b * CF * NPTS;
  stage_panel128(tb_lds, fb + nb);
  auto ga = [&](int m, int k) { return W[(mb + m)*CF + k]; };
  auto gb = [&](int k, int nl) { return tb_lds[k*17 + nl]; };
  v8f acc = wmma_loop<4>(ga, gb);
  const int lane = threadIdx.x & 31, half = lane >> 4, nl = lane & 15;
  float vals[8];
  #pragma unroll
  for (int r = 0; r < 8; ++r) {
    const int M = r + half*8;
    const float v = acc[r] + bias[mb + M];
    out[(size_t)b*CO*NPTS + (size_t)(mb + M)*NPTS + (nb + nl)] = v;
    vals[r] = v;
  }
  group_accum(gsum, gss, b, mtile, vals, half, lane);
}

// ---------------- groupnorm stats finalize ----------------
__global__ void k_gn_fin(const float* gsum, const float* gss, float* st) {
  const int i = threadIdx.x;
  if (i < BATCH * NG) {
    const float mu = gsum[i] * INVCNT;
    const float var = gss[i] * INVCNT - mu*mu;
    st[2*i]   = mu;
    st[2*i+1] = rsqrtf(var + GEPS);
  }
}

// ---------------- feat = relu(gn2(pre2)) + relu(gn3(pre3)) ----------------
__global__ void k_feat(const float* pre2, const float* st2, const float* g2, const float* bb2,
                       const float* pre3, const float* st3, const float* g3, const float* bb3,
                       float* feat) {
  const size_t i = (size_t)blockIdx.x * blockDim.x + threadIdx.x;
  const int b = (int)(i / ((size_t)CO * NPTS));
  const int c = (int)((i / NPTS) % CO);
  const int s = (b*NG + (c >> 2)) * 2;
  const float h = fmaxf((pre2[i] - st2[s]) * st2[s+1] * g2[c] + bb2[c], 0.f);
  const float r = fmaxf((pre3[i] - st3[s]) * st3[s+1] * g3[c] + bb3[c], 0.f);
  feat[i] = h + r;
}

// ---------------- q GEMM: q[b][n][o] = q_w @ feat[:,fidx] + q_b ----------------
__global__ void k_q_gemm(const float* qw, const float* qb,
                         const float* feat, const int* fidx, float* q) {
  const int mb = blockIdx.x * 16, ob = blockIdx.y * 16, b = blockIdx.z;
  const float* fb = feat + (size_t)b * CO * NPTS;
  const int* fi = fidx + b * NP;
  auto ga = [&](int m, int k) { return fb[(size_t)k*NPTS + fi[mb + m]]; };
  auto gb = [&](int k, int nl) { return qw[(ob + nl)*CO + k]; };
  v8f acc = wmma_loop<4>(ga, gb);
  const int lane = threadIdx.x & 31, half = lane >> 4, nl = lane & 15;
  #pragma unroll
  for (int r = 0; r < 8; ++r) {
    const int M = r + half*8;
    q[((size_t)b*NP + (mb + M))*CO + (ob + nl)] = acc[r] + qb[ob + nl];
  }
}

// ---------------- qkw GEMM: (q^T K_w)*scale, col 137 = (q . k_b)*scale ---------
__global__ void k_qkw_gemm(const float* q, const float* kw, const float* kb,
                           float* qkw) {
  const int mb = blockIdx.x * 16, cb = blockIdx.y * 16, b = blockIdx.z;
  const float* qrow = q + (size_t)b * NP * CO;
  auto ga = [&](int m, int k) { return qrow[(size_t)(mb + m)*CO + k]; };
  auto gb = [&](int k, int nl) {
    const int cg = cb + nl;
    if (cg < CG) return kw[k*CG + cg];
    if (cg == CG) return kb[k];
    return 0.f;
  };
  v8f acc = wmma_loop<4>(ga, gb);
  const int lane = threadIdx.x & 31, half = lane >> 4, nl = lane & 15;
  #pragma unroll
  for (int r = 0; r < 8; ++r) {
    const int M = r + half*8;
    qkw[((size_t)b*NP + (mb + M))*CGP + (cb + nl)] = acc[r] * QKSCALE;
  }
}

// ---------------- V GEMM: v_w @ grouped (137->160 padded K), gather on the fly -
__global__ void k_v_gemm(const float* vw, const float* vb,
                         const float* feat, const int* idxb,
                         const float* xyz, const float* newxyz,
                         float* vpre, float* gsum, float* gss) {
  const int nb = blockIdx.x * 16, mtile = blockIdx.y, b = blockIdx.z;
  const int mb = mtile * 16;
  const float* fb = feat + (size_t)b * CO * NPTS;
  const float* xb = xyz + (size_t)b * NPTS * 3;
  auto ga = [&](int m, int k) { return (k < CG) ? vw[(mb + m)*CG + k] : 0.f; };
  auto gb = [&](int r, int nl) {
    const int col = nb + nl;
    const int n = col >> 5, kk = col & 31;
    if (r >= CG) return 0.f;
    const int pid = idxb[((size_t)(b*NP + n))*KNN + kk];
    if (r >= 9) return fb[(size_t)(r - 9)*NPTS + pid];
    const float* cen = newxyz + ((size_t)b*NP + n)*3;
    if (r < 3) return xb[pid*3 + r];
    if (r < 6) return xb[pid*3 + (r - 3)] - cen[r - 3];
    return cen[r - 6];
  };
  v8f acc = wmma_loop<5>(ga, gb);
  const int lane = threadIdx.x & 31, half = lane >> 4, nl = lane & 15;
  float vals[8];
  #pragma unroll
  for (int r = 0; r < 8; ++r) {
    const int M = r + half*8;
    const float v = acc[r] + vb[mb + M];
    vpre[(size_t)b*CO*NP*KNN + (size_t)(mb + M)*NP*KNN + (nb + nl)] = v;
    vals[r] = v;
  }
  group_accum(gsum, gss, b, mtile, vals, half, lane);
}

// ---------------- attention: qk dot, masked softmax, weighted gn(v) ------------
__global__ void k_attn(const float* qkw, const float* feat, const int* idxb,
                       const int* cntb, const float* xyz, const float* newxyz,
                       const float* vpre, const float* stv,
                       const float* vg, const float* vbb, float* outfeat) {
  const int ci = blockIdx.x, b = ci >> 9, n = ci & 511;
  __shared__ float sattn[KNN];
  const float* qw = qkw + ((size_t)b*NP + n)*CGP;
  const int tid = threadIdx.x;
  if (tid < 32) {
    const int k = tid;
    const int pid = idxb[((size_t)(b*NP + n))*KNN + k];
    const float* cen = newxyz + ((size_t)b*NP + n)*3;
    const float* gp = xyz + ((size_t)b*NPTS + pid)*3;
    float geo[9] = { gp[0], gp[1], gp[2],
                     gp[0]-cen[0], gp[1]-cen[1], gp[2]-cen[2],
                     cen[0], cen[1], cen[2] };
    float qk = qw[CG];   // (q . k_b) * scale term
    #pragma unroll
    for (int c = 0; c < 9; ++c) qk += qw[c] * geo[c];
    const float* fb = feat + (size_t)b * CO * NPTS;
    for (int c = 0; c < CO; ++c) qk += qw[9 + c] * fb[(size_t)c*NPTS + pid];
    const int cc = max(cntb[b*NP + n], 1);
    const bool valid = k < cc;
    float qv = valid ? qk : -1e9f;
    float m = qv;
    #pragma unroll
    for (int o = 16; o > 0; o >>= 1) m = fmaxf(m, __shfl_xor(m, o, 32));
    float e = valid ? expf(qv - m) : 0.f;
    float s = e;
    #pragma unroll
    for (int o = 16; o > 0; o >>= 1) s += __shfl_xor(s, o, 32);
    sattn[k] = e / s;
  }
  __syncthreads();
  {
    const int c = tid;                          // block of 128 threads
    const int s = (b*NG + (c >> 2)) * 2;
    const float mu = stv[s], rs = stv[s+1], gg = vg[c], be = vbb[c];
    const float* vp = vpre + (size_t)b*CO*NP*KNN + (size_t)c*NP*KNN + (size_t)n*KNN;
    float acc = 0.f;
    #pragma unroll
    for (int k = 0; k < KNN; ++k) {
      const float y = fmaxf((vp[k] - mu) * rs * gg + be, 0.f);
      acc += sattn[k] * y;
    }
    outfeat[(size_t)b*CO*NP + (size_t)c*NP + n] = (float)KNN * acc;
  }
}

// ---------------- host launcher ----------------
extern "C" void kernel_launch(void* const* d_in, const int* in_sizes, int n_in,
                              void* d_out, int out_size, void* d_ws, size_t ws_size,
                              hipStream_t stream) {
  const float* xyz      = (const float*)d_in[0];
  const float* features = (const float*)d_in[1];
  const float* t_emb    = (const float*)d_in[2];
  const float* cond_emb = (const float*)d_in[3];
  const float* t_w1 = (const float*)d_in[4];  const float* t_b1 = (const float*)d_in[5];
  const float* t_w2 = (const float*)d_in[6];  const float* t_b2 = (const float*)d_in[7];
  const float* c_w1 = (const float*)d_in[8];  const float* c_b1 = (const float*)d_in[9];
  const float* c_w2 = (const float*)d_in[10]; const float* c_b2 = (const float*)d_in[11];
  const float* fm_w = (const float*)d_in[12]; const float* fm_b = (const float*)d_in[13];
  const float* fm_g = (const float*)d_in[14]; const float* fm_bb= (const float*)d_in[15];
  const float* sm_w = (const float*)d_in[16]; const float* sm_b = (const float*)d_in[17];
  const float* sm_g = (const float*)d_in[18]; const float* sm_bb= (const float*)d_in[19];
  const float* rc_w = (const float*)d_in[20]; const float* rc_b = (const float*)d_in[21];
  const float* rc_g = (const float*)d_in[22]; const float* rc_bb= (const float*)d_in[23];
  const float* q_w  = (const float*)d_in[24]; const float* q_b  = (const float*)d_in[25];
  const float* k_w  = (const float*)d_in[26]; const float* k_b  = (const float*)d_in[27];
  const float* v_w  = (const float*)d_in[28]; const float* v_b  = (const float*)d_in[29];
  const float* v_g  = (const float*)d_in[30]; const float* v_bb = (const float*)d_in[31];

  float* out_xyz  = (float*)d_out;                       // (B, NP, 3)
  float* out_feat = out_xyz + (size_t)BATCH*NP*3;        // (B, CO, NP)

  // workspace layout (floats); feat aliases pre1, vpre aliases pre2
  float* f = (float*)d_ws;
  const size_t SZ1 = (size_t)BATCH * CO * NPTS;          // 16,777,216
  float* pre1 = f;
  float* pre2 = f + SZ1;
  float* pre3 = f + 2*SZ1;
  float* feat = pre1;                                    // reuse after sm GEMM
  float* vpre = pre2;                                    // reuse after k_feat
  size_t S = 3*SZ1;
  float* q    = f + S; S += (size_t)BATCH*NP*CO;
  float* qkw  = f + S; S += (size_t)BATCH*NP*CGP;
  float* temb = f + S; S += BATCH*128;
  float* cemb = f + S; S += BATCH*128;
  float* st1  = f + S; S += BATCH*NG*2;
  float* st2  = f + S; S += BATCH*NG*2;
  float* st3  = f + S; S += BATCH*NG*2;
  float* stv  = f + S; S += BATCH*NG*2;
  float* acc0 = f + S;                                   // 8 x 256 accumulators
  float* gsum1 = acc0;        float* gss1 = acc0 + 256;
  float* gsum2 = acc0 + 512;  float* gss2 = acc0 + 768;
  float* gsum3 = acc0 + 1024; float* gss3 = acc0 + 1280;
  float* gsumv = acc0 + 1536; float* gssv = acc0 + 1792;
  S += 2048;
  int* ip   = (int*)(f + S);
  int* fidx = ip;                                        // B*NP
  int* idxb = ip + BATCH*NP;                             // B*NP*KNN
  int* cntb = idxb + (size_t)BATCH*NP*KNN;               // B*NP

  (void)hipMemsetAsync(acc0, 0, 2048 * sizeof(float), stream);

  k_embed<<<BATCH, 128, 0, stream>>>(t_emb, t_w1, t_b1, t_w2, t_b2,
                                     cond_emb, c_w1, c_b1, c_w2, c_b2,
                                     temb, cemb);
  k_fps<<<BATCH, 512, 0, stream>>>(xyz, fidx, out_xyz);
  k_ball<<<BATCH*NP/8, 256, 0, stream>>>(xyz, out_xyz, idxb, cntb);

  dim3 ggrid(NPTS/16, CO/16, BATCH);
  k_gemm_fm<<<ggrid, 32, 0, stream>>>(fm_w, fm_b, features, temb, pre1, gsum1, gss1);
  k_gn_fin<<<1, 256, 0, stream>>>(gsum1, gss1, st1);
  k_gemm_sm<<<ggrid, 32, 0, stream>>>(sm_w, sm_b, pre1, st1, fm_g, fm_bb, cemb,
                                      pre2, gsum2, gss2);
  k_gn_fin<<<1, 256, 0, stream>>>(gsum2, gss2, st2);
  k_gemm_rc<<<ggrid, 32, 0, stream>>>(rc_w, rc_b, features, pre3, gsum3, gss3);
  k_gn_fin<<<1, 256, 0, stream>>>(gsum3, gss3, st3);
  k_feat<<<(unsigned)((size_t)BATCH*CO*NPTS/256), 256, 0, stream>>>(
      pre2, st2, sm_g, sm_bb, pre3, st3, rc_g, rc_bb, feat);

  k_q_gemm<<<dim3(NP/16, CO/16, BATCH), 32, 0, stream>>>(q_w, q_b, feat, fidx, q);
  k_qkw_gemm<<<dim3(NP/16, CGP/16, BATCH), 32, 0, stream>>>(q, k_w, k_b, qkw);
  k_v_gemm<<<dim3(NP*KNN/16, CO/16, BATCH), 32, 0, stream>>>(
      v_w, v_b, feat, idxb, xyz, out_xyz, vpre, gsumv, gssv);
  k_gn_fin<<<1, 256, 0, stream>>>(gsumv, gssv, stv);
  k_attn<<<BATCH*NP, 128, 0, stream>>>(qkw, feat, idxb, cntb, xyz, out_xyz,
                                       vpre, stv, v_g, v_bb, out_feat);
}
